// WindowAttention_9045201125807
// MI455X (gfx1250) — matrix operand order repaired and checked by reading, not compiled, not run on hardware
//
#include <hip/hip_runtime.h>
#include <hip/hip_bf16.h>

typedef __bf16 bf16_t;
typedef __attribute__((ext_vector_type(16))) __bf16 v16bf;
typedef __attribute__((ext_vector_type(8)))  float  v8f;
typedef __attribute__((ext_vector_type(4)))  float  v4f;
typedef __attribute__((ext_vector_type(4)))  __bf16 v4bf;

union FragU { v16bf v; unsigned int u[8]; };

#define NTOK 98
#define NPAD 128
#define SCALE_F 0.17677669529663687f

// ---- gfx1250 async copy: global -> LDS, 16 bytes per active lane ----------
// Tracked by ASYNCcnt; no VGPR data round-trip (CDNA5 ISA).
__device__ __forceinline__ void async_copy16(unsigned lds_addr, const void* g) {
  asm volatile("global_load_async_to_lds_b128 %0, %1, off"
               :: "v"(lds_addr), "v"(g)
               : "memory");
}
__device__ __forceinline__ void wait_async0() {
  asm volatile("s_wait_asynccnt 0x0" ::: "memory");
}
// Generic (flat) pointer to a __shared__ object: low 32 bits are the LDS byte
// address (LDS aperture maps addr[31:0] -> LDS offset).
__device__ __forceinline__ unsigned lds_addr_of(const void* p) {
  return (unsigned)(uintptr_t)p;
}

// ---- WMMA 16-bit fragment helpers (wave32), contiguous k order k = 16h + s (not the ISA table order) ----
// A-matrix 16x32 (MxK), order used here: lanes 0-15 row M=lane; VGPR v holds K pair {2v,2v+1}
// (K=0..15); lanes 16-31 same rows, K offset +16.
__device__ __forceinline__ int ka0(int lane, int v) {
  int h16 = (lane >> 4) << 4;                     // 0 or 16
  return h16 + (v << 1);
}
__device__ __forceinline__ v16bf frag_a32(const unsigned int* rowdw, int lane) {
  FragU f;
#pragma unroll
  for (int v = 0; v < 8; ++v) f.u[v] = rowdw[ka0(lane, v) >> 1];
  return f.v;
}
// B-matrix 32x16 (KxN): lane = column N (mod 16); same contiguous order as A: lanes 0-15 hold K=0..15,
// lanes 16-31 hold K=16..31, VGPR v = K pair {2v,2v+1}. From a row-major
// [token][32] source this is 8 contiguous dwords per lane-half.
__device__ __forceinline__ v16bf frag_b32(const unsigned int* rowdw, int lane) {
  FragU f; int off = (lane >> 4) << 3;
#pragma unroll
  for (int v = 0; v < 8; ++v) f.u[v] = rowdw[off + v];
  return f.v;
}

// ---------------------------------------------------------------------------
// Kernel 0: vectorized fp32 -> bf16 conversion (n must be a multiple of 4).
// ---------------------------------------------------------------------------
__global__ __launch_bounds__(256)
void f32_to_bf16_kernel(const float* __restrict__ src, bf16_t* __restrict__ dst,
                        long n4) {
  long i = (long)blockIdx.x * blockDim.x + threadIdx.x;
  long stride = (long)gridDim.x * blockDim.x;
  const v4f* s4 = (const v4f*)src;
  v4bf* d4 = (v4bf*)dst;
  for (; i < n4; i += stride)
    d4[i] = __builtin_convertvector(s4[i], v4bf);
}

// ---------------------------------------------------------------------------
// Kernel 1: expand relative-position bias to biasT[h][q][k]  (24 x 98 x 98 f32)
// ---------------------------------------------------------------------------
__global__ void bias_expand_kernel(const float* __restrict__ table,
                                   const int* __restrict__ index,
                                   float* __restrict__ biasT) {
  int t = blockIdx.x * blockDim.x + threadIdx.x;
  if (t >= NTOK * NTOK) return;
  int idx = index[t];
#pragma unroll
  for (int hh = 0; hh < 24; ++hh)
    biasT[(long)hh * (NTOK * NTOK) + t] = table[idx * 24 + hh];
}

// ---------------------------------------------------------------------------
// Kernel 2: QKV GEMM (25088 x 2304 = x @ W^T + b), bf16 WMMA, f32 accumulate.
// Double-buffered LDS staged via global_load_async_to_lds_b128 (ASYNCcnt).
// Epilogue decodes j = nh*96 + hd*3 + which, folds softmax SCALE into Q,
// stores bf16 into head-major [b*24+nh][tok][hd] workspaces.
// ---------------------------------------------------------------------------
__global__ __launch_bounds__(256)
void qkv_gemm_kernel(const bf16_t* __restrict__ xb, const bf16_t* __restrict__ wb,
                     const float* __restrict__ bias,
                     bf16_t* __restrict__ qb, bf16_t* __restrict__ kb,
                     bf16_t* __restrict__ vb) {
  __shared__ __align__(16) bf16_t sA[2][128 * 32];
  __shared__ __align__(16) bf16_t sB[2][64 * 32];
  const int tid  = threadIdx.x;
  const int lane = tid & 31;
  const int wid  = tid >> 5;
  const int wm = wid & 3, wn = wid >> 2;      // 4x2 wave grid, 32x32 per wave
  const int jt = blockIdx.x;                  // 0..35
  const int mt = blockIdx.y;                  // 0..195
  const long m0 = (long)mt * 128;
  const int  j0 = jt * 64;

  const v8f vzero = {0.f,0.f,0.f,0.f,0.f,0.f,0.f,0.f};
  v8f acc[2][2];
#pragma unroll
  for (int a = 0; a < 2; ++a)
#pragma unroll
    for (int c = 0; c < 2; ++c) acc[a][c] = vzero;

  // Stage tile kk into LDS buffer `buf` with async DMA (16B per lane-chunk).
  auto stage = [&](int buf, int kk) {
    const int k0 = kk * 32;
    unsigned aBase = lds_addr_of(&sA[buf][0]);
    unsigned bBase = lds_addr_of(&sB[buf][0]);
    // A tile: 128 rows x 64B = 512 chunks, 2 per thread
#pragma unroll
    for (int it = 0; it < 2; ++it) {
      int i = tid + it * 256;
      int r = i >> 2, c = i & 3;              // c: 16B sub-chunk in row
      async_copy16(aBase + (unsigned)(r * 64 + c * 16),
                   xb + (m0 + r) * 768 + k0 + c * 8);
    }
    // B tile: 64 rows x 64B = 256 chunks, 1 per thread
    {
      int r = tid >> 2, c = tid & 3;
      async_copy16(bBase + (unsigned)(tid * 16),
                   wb + (long)(j0 + r) * 768 + k0 + c * 8);
    }
  };

  stage(0, 0);
  wait_async0();
  __syncthreads();

  for (int kk = 0; kk < 24; ++kk) {
    const int cur = kk & 1;
    if (kk + 1 < 24) stage(cur ^ 1, kk + 1);          // overlap DMA w/ compute
    if (kk + 2 < 24)                                  // L2 prefetch, tile+2
      __builtin_prefetch(xb + (m0 + (tid >> 1)) * 768 + (kk + 2) * 32, 0, 1);

    v16bf am[2], bn[2];
#pragma unroll
    for (int f = 0; f < 2; ++f) {
      int arow = wm * 32 + f * 16 + (lane & 15);
      am[f] = frag_a32((const unsigned int*)(&sA[cur][0] + arow * 32), lane);
      int bcol = wn * 32 + f * 16 + (lane & 15);
      bn[f] = frag_b32((const unsigned int*)(&sB[cur][0] + bcol * 32), lane);
    }
#pragma unroll
    for (int fa = 0; fa < 2; ++fa)
#pragma unroll
      for (int fb = 0; fb < 2; ++fb)
        acc[fa][fb] = __builtin_amdgcn_wmma_f32_16x16x32_bf16(
            false, am[fa], false, bn[fb], (short)0, acc[fa][fb], false, false);

    wait_async0();                                    // next tile landed
    __syncthreads();                                  // all waves done reading
  }

  // Epilogue: C layout row = vgpr + 8*(lane>>4), col = lane&15.
  const int rowhalf = (lane >> 4) << 3;
  const int col_lo  = lane & 15;
#pragma unroll
  for (int fa = 0; fa < 2; ++fa) {
#pragma unroll
    for (int fb = 0; fb < 2; ++fb) {
      int j = j0 + wn * 32 + fb * 16 + col_lo;
      int which = j % 3;
      int rest  = j / 3;
      int hd = rest & 31;
      int nh = rest >> 5;
      bf16_t* dst = (which == 0) ? qb : ((which == 1) ? kb : vb);
      float scl = (which == 0) ? SCALE_F : 1.0f;
      float bj  = bias[j];
#pragma unroll
      for (int v = 0; v < 8; ++v) {
        long m = m0 + wm * 32 + fa * 16 + rowhalf + v;
        int bwin = (int)(m / NTOK);
        int tok  = (int)(m - (long)bwin * NTOK);
        float val = (acc[fa][fb][v] + bj) * scl;
        dst[(((long)bwin * 24 + nh) * NTOK + tok) * 32 + hd] = (bf16_t)val;
      }
    }
  }
}

// ---------------------------------------------------------------------------
// Kernel 3: per-(window,head) attention. Block = 4 waves; K/V async-staged to
// LDS (zero padded to 128 tokens); each wave owns q-tiles (7 tiles of 16 rows).
// QK^T: 8 WMMAs (K=HD=32); softmax in-register via __shfl_xor over 16-lane
// column groups; P -> bf16 per-wave LDS tile; PV: 2x4 WMMAs; f32 stores.
// ---------------------------------------------------------------------------
__global__ __launch_bounds__(128)
void attn_kernel(const bf16_t* __restrict__ qb, const bf16_t* __restrict__ kb,
                 const bf16_t* __restrict__ vb, const float* __restrict__ biasT,
                 const float* __restrict__ mask, float* __restrict__ out) {
  __shared__ __align__(16) bf16_t sK[NPAD * 32];
  __shared__ __align__(16) bf16_t sV[NPAD * 32];
  __shared__ __align__(16) bf16_t sP[4][16 * NPAD];
  const int h = blockIdx.x;       // 0..23
  const int b = blockIdx.y;       // 0..255
  const int tid  = threadIdx.x;
  const int lane = tid & 31;
  const int wid  = tid >> 5;
  const long bh = (long)b * 24 + h;

  { // stage K,V: 98 rows (6272B = 392 x 16B chunks) via async DMA; pad -> 0
    const char* kg = (const char*)(kb + bh * NTOK * 32);
    const char* vg = (const char*)(vb + bh * NTOK * 32);
    unsigned sK0 = lds_addr_of(sK);
    unsigned sV0 = lds_addr_of(sV);
    for (int i = tid; i < 392; i += 128) {
      async_copy16(sK0 + (unsigned)(i * 16), kg + i * 16);
      async_copy16(sV0 + (unsigned)(i * 16), vg + i * 16);
    }
    unsigned int* sKd = (unsigned int*)sK;
    unsigned int* sVd = (unsigned int*)sV;
    for (int i = tid; i < 480; i += 128) {    // rows 98..127 zeroed
      sKd[1568 + i] = 0u;
      sVd[1568 + i] = 0u;
    }
    wait_async0();
  }
  __syncthreads();

  const int col_lo  = lane & 15;
  const int rowhalf = (lane >> 4) << 3;
  const float* biasH = biasT + (long)h * (NTOK * NTOK);
  const float* maskW = mask  + (long)(b & 63) * (NTOK * NTOK);
  const v8f vzero = {0.f,0.f,0.f,0.f,0.f,0.f,0.f,0.f};

  // Hoist V B-fragments: B element (K=kt, N=d) = V[kt][d]; lane = d column.
  v16bf bv[2][4];
  const unsigned short* sVs = (const unsigned short*)sV;
#pragma unroll
  for (int nt = 0; nt < 2; ++nt) {
#pragma unroll
    for (int kc = 0; kc < 4; ++kc) {
      FragU f;
      int half16 = (lane >> 4) << 4;
      int colv = nt * 16 + col_lo;
#pragma unroll
      for (int v = 0; v < 8; ++v) {
        int kkr = kc * 32 + half16 + (v << 1);
        unsigned int lo = sVs[kkr * 32 + colv];
        unsigned int hi = sVs[(kkr + 1) * 32 + colv];
        f.u[v] = lo | (hi << 16);
      }
      bv[nt][kc] = f.v;
    }
  }
  // Hoist K B-fragments (contiguous 32B row reads).
  v16bf bk[8];
#pragma unroll
  for (int kt = 0; kt < 8; ++kt)
    bk[kt] = frag_b32((const unsigned int*)(sK + (kt * 16 + col_lo) * 32), lane);

  for (int qt = wid; qt < 7; qt += 4) {
    int qrow = qt * 16 + col_lo;
    if (qrow > 97) qrow = 97;                 // clamp; padded rows discarded
    v16bf aq = frag_a32((const unsigned int*)(qb + (bh * NTOK + qrow) * 32), lane);

    v8f s[8];
#pragma unroll
    for (int kt = 0; kt < 8; ++kt)
      s[kt] = __builtin_amdgcn_wmma_f32_16x16x32_bf16(
          false, aq, false, bk[kt], (short)0, vzero, false, false);

    float mx[8];
#pragma unroll
    for (int v = 0; v < 8; ++v) mx[v] = -3.0e38f;
#pragma unroll
    for (int kt = 0; kt < 8; ++kt) {
      int c = kt * 16 + col_lo;
#pragma unroll
      for (int v = 0; v < 8; ++v) {
        int r = qt * 16 + rowhalf + v;
        if (r > 97) r = 97;
        if (c < 98) s[kt][v] += biasH[r * NTOK + c] + maskW[r * NTOK + c];
        else        s[kt][v] = -3.0e38f;      // pad columns -> exp = 0
        mx[v] = fmaxf(mx[v], s[kt][v]);
      }
    }
#pragma unroll
    for (int v = 0; v < 8; ++v)
#pragma unroll
      for (int off = 1; off < 16; off <<= 1)
        mx[v] = fmaxf(mx[v], __shfl_xor(mx[v], off, 32));

    float sm[8];
#pragma unroll
    for (int v = 0; v < 8; ++v) sm[v] = 0.f;
#pragma unroll
    for (int kt = 0; kt < 8; ++kt)
#pragma unroll
      for (int v = 0; v < 8; ++v) {
        float e = __expf(s[kt][v] - mx[v]);
        s[kt][v] = e;
        sm[v] += e;
      }
#pragma unroll
    for (int v = 0; v < 8; ++v) {
#pragma unroll
      for (int off = 1; off < 16; off <<= 1)
        sm[v] += __shfl_xor(sm[v], off, 32);
      sm[v] = 1.0f / sm[v];
    }

    // P tile (16 x 128 bf16) to per-wave LDS; C layout -> row-major scatter.
    bf16_t* P = sP[wid];
#pragma unroll
    for (int kt = 0; kt < 8; ++kt) {
      int c = kt * 16 + col_lo;
#pragma unroll
      for (int v = 0; v < 8; ++v)
        P[(rowhalf + v) * NPAD + c] = (bf16_t)(s[kt][v] * sm[v]);
    }
    __builtin_amdgcn_wave_barrier();          // per-wave LDS; HW keeps DS in order
    asm volatile("" ::: "memory");

    // O = P @ V
    const unsigned int* prow = (const unsigned int*)(P + col_lo * NPAD);
#pragma unroll
    for (int nt = 0; nt < 2; ++nt) {
      v8f o = vzero;
#pragma unroll
      for (int kc = 0; kc < 4; ++kc) {
        FragU f;
#pragma unroll
        for (int v = 0; v < 8; ++v)
          f.u[v] = prow[(kc * 32 + ka0(lane, v)) >> 1];
        o = __builtin_amdgcn_wmma_f32_16x16x32_bf16(
            false, f.v, false, bv[nt][kc], (short)0, o, false, false);
      }
#pragma unroll
      for (int v = 0; v < 8; ++v) {
        int r = qt * 16 + rowhalf + v;
        if (r < 98)
          out[((long)b * NTOK + r) * 768 + h * 32 + nt * 16 + col_lo] = o[v];
      }
    }
    __builtin_amdgcn_wave_barrier();
  }
}

// ---------------------------------------------------------------------------
extern "C" void kernel_launch(void* const* d_in, const int* in_sizes, int n_in,
                              void* d_out, int out_size, void* d_ws, size_t ws_size,
                              hipStream_t stream) {
  const float* x     = (const float*)d_in[0];
  const float* mask  = (const float*)d_in[1];
  const float* w     = (const float*)d_in[2];
  const float* bias  = (const float*)d_in[3];
  const float* table = (const float*)d_in[4];
  const int*   ridx  = (const int*)d_in[5];
  float* out = (float*)d_out;

  char* ws = (char*)d_ws;
  const size_t Sx = (size_t)25088 * 768 * sizeof(bf16_t);          // 38.5 MB
  const size_t Sw = (size_t)2304 * 768 * sizeof(bf16_t);           // 3.5 MB
  const size_t S1 = (size_t)256 * 24 * NTOK * 32 * sizeof(bf16_t); // 38.5 MB
  bf16_t* xb = (bf16_t*)(ws);
  bf16_t* wb = (bf16_t*)(ws + Sx);
  bf16_t* qb = (bf16_t*)(ws + Sx + Sw);
  bf16_t* kb = (bf16_t*)(ws + Sx + Sw + S1);
  bf16_t* vb = (bf16_t*)(ws + Sx + Sw + 2 * S1);
  float* biasT = (float*)(ws + Sx + Sw + 3 * S1);                  // 24*98*98 f32

  f32_to_bf16_kernel<<<1024, 256, 0, stream>>>(x, xb, (long)(25088 * 768) / 4);
  f32_to_bf16_kernel<<<256, 256, 0, stream>>>(w, wb, (long)(2304 * 768) / 4);
  bias_expand_kernel<<<(NTOK * NTOK + 127) / 128, 128, 0, stream>>>(table, ridx, biasT);
  qkv_gemm_kernel<<<dim3(36, 196), 256, 0, stream>>>(xb, wb, bias, qb, kb, vb);
  attn_kernel<<<dim3(24, 256), 128, 0, stream>>>(qb, kb, vb, biasT, mask, out);
}
